// Attention_47115791237601
// MI455X (gfx1250) — matrix operand order repaired and checked by reading, not compiled, not run on hardware
//
#include <hip/hip_runtime.h>
#include <hip/hip_bf16.h>

// ---------------------------------------------------------------------------
// Types for CDNA5 WMMA (wave32): A/B = 16 bf16 per lane (8 VGPRs), C/D = 8 f32
// ---------------------------------------------------------------------------
typedef __attribute__((ext_vector_type(16))) __bf16 v16bf;
typedef __attribute__((ext_vector_type(8)))  float  v8f;

#define N_TOK 4096   // 64*64 tokens
#define DH    32     // dim per head
#define NH    4      // heads
#define NB    4      // batch
#define CCH   128    // channels

// SCALE * log2(e): fold the softmax base-2 conversion into the Q pre-scale
#define QSCALE (0.17677669529663687f * 1.4426950408889634f)

__device__ __forceinline__ v8f wmma_bf16(v16bf a, v16bf b, v8f c) {
    return __builtin_amdgcn_wmma_f32_16x16x32_bf16(false, a, false, b, (short)0, c, false, false);
}

// 16 contiguous bf16 (32B) -> v16bf  (lane holds 16 consecutive K; unused, operands use load16_split)
__device__ __forceinline__ v16bf load16_contig(const __bf16* p) {
    union { v16bf v; uint4 u[2]; } r;
    r.u[0] = *(const uint4*)(p);
    r.u[1] = *(const uint4*)(p + 8);
    return r.v;
}
// A-operand: lane holds 8 bf16 at p and 8 bf16 at p+16
__device__ __forceinline__ v16bf load16_split(const __bf16* p) {
    union { v16bf v; uint4 u[2]; } r;
    r.u[0] = *(const uint4*)(p);
    r.u[1] = *(const uint4*)(p + 16);
    return r.v;
}

// ---------------------------------------------------------------------------
// Stage 1: QKV 1x1-conv projection.
//   Q : [b][head][n][32] bf16, pre-scaled by dh^-0.5 * log2(e)
//   K : [b][head][n][32] bf16   (token-major: A-operand loads contiguous)
//   Vt: [b][head][32][n] bf16   (d-major: P.V B-operand loads contiguous)
// ---------------------------------------------------------------------------
__global__ void __launch_bounds__(256) qkv_kernel(
    const float* __restrict__ x, const float* __restrict__ w,
    __bf16* __restrict__ Q, __bf16* __restrict__ K, __bf16* __restrict__ Vt)
{
    const int i = blockIdx.x * 256 + threadIdx.x;   // token (coalesced)
    const int o = blockIdx.y;                       // output row 0..383
    const int b = blockIdx.z;
    const float* xb = x + (size_t)b * CCH * N_TOK + i;
    const float* wr = w + o * CCH;                  // uniform -> scalar loads
    float acc = 0.f;
    #pragma unroll 8
    for (int c = 0; c < CCH; ++c)
        acc = fmaf(wr[c], xb[(size_t)c * N_TOK], acc);

    const int which = o >> 7;        // 0=q 1=k 2=v
    const int oo    = o & 127;
    const int head  = oo >> 5, d = oo & 31;
    const size_t bh = (size_t)b * NH + head;
    if (which == 0)
        Q[(bh * N_TOK + i) * DH + d] = (__bf16)(acc * QSCALE);
    else if (which == 1)
        K[(bh * N_TOK + i) * DH + d] = (__bf16)acc;
    else
        Vt[(bh * DH + d) * N_TOK + i] = (__bf16)acc;   // coalesced over i
}

// ---------------------------------------------------------------------------
// Stage 2: flash attention, one wave per 16-query tile, 64 keys per step.
//   Computes S^T = K_tile x Q^T so the softmax reduction is in-lane and the
//   C-layout of S^T *is* the A-layout of P for the P.V WMMA (no LDS bounce).
//   Hid: [b][head*32+d][n] f32
// ---------------------------------------------------------------------------
__global__ void __launch_bounds__(256) flash_attn_kernel(
    const __bf16* __restrict__ Q, const __bf16* __restrict__ K,
    const __bf16* __restrict__ Vt, float* __restrict__ Hid)
{
    const int lane = threadIdx.x & 31;
    const int wv   = threadIdx.x >> 5;
    const int wid  = blockIdx.x * 8 + wv;
    const int qtile = wid & 255;                    // 256 query tiles of 16
    const int bh    = wid >> 8;                     // 0..15
    const int b = bh >> 2, head = bh & 3;

    const __bf16* Qbh = Q  + (size_t)bh * N_TOK * DH;
    const __bf16* Kbh = K  + (size_t)bh * N_TOK * DH;
    const __bf16* Vbh = Vt + (size_t)bh * DH * N_TOK;

    const int m     = lane & 15;
    const int hi    = lane >> 4;
    const int base0 = hi * 8;
    const int rowln = hi * 8;                       // source-lane base for row bcasts

    // Stationary B operand: Q^T (32d x 16q). lane: query col m, d hi*8..+8 and 16+hi*8..+8
    const v16bf bQ = load16_split(Qbh + (size_t)(qtile * 16 + m) * DH + hi * 8);

    v8f O0 = {}, O1 = {};                           // O tile: rows=query, cols=d 0..15/16..31
    float mrun = -1e30f, lrun = 0.f;                // per-query (per-lane) running stats

    for (int j0 = 0; j0 < N_TOK; j0 += 64) {
        // A = K token tiles (16 keys x 32 d)
        const v16bf aK0 = load16_split(Kbh + (size_t)(j0 +  0 + m) * DH + base0);
        const v16bf aK1 = load16_split(Kbh + (size_t)(j0 + 16 + m) * DH + base0);
        const v16bf aK2 = load16_split(Kbh + (size_t)(j0 + 32 + m) * DH + base0);
        const v16bf aK3 = load16_split(Kbh + (size_t)(j0 + 48 + m) * DH + base0);
        v8f z0 = {}, z1 = {}, z2 = {}, z3 = {};
        v8f St0 = wmma_bf16(aK0, bQ, z0);           // S^T: rows=key, cols=query
        v8f St1 = wmma_bf16(aK1, bQ, z1);
        v8f St2 = wmma_bf16(aK2, bQ, z2);
        v8f St3 = wmma_bf16(aK3, bQ, z3);

        // B = V tiles (32k x 16d): lane: d col, keys hi*8..+8 and 16+hi*8..+8 (split)
        const v16bf bV00 = load16_split(Vbh + (size_t)(m)      * N_TOK + j0      + hi * 8);
        const v16bf bV01 = load16_split(Vbh + (size_t)(m)      * N_TOK + j0 + 32 + hi * 8);
        const v16bf bV10 = load16_split(Vbh + (size_t)(m + 16) * N_TOK + j0      + hi * 8);
        const v16bf bV11 = load16_split(Vbh + (size_t)(m + 16) * N_TOK + j0 + 32 + hi * 8);

        // per-query max over the 32 in-lane S^T values (in-lane VALU tree)
        float mx = -1e30f;
        #pragma unroll
        for (int v = 0; v < 8; ++v)
            mx = fmaxf(mx, fmaxf(fmaxf(St0[v], St1[v]), fmaxf(St2[v], St3[v])));
        mx = fmaxf(mx, __shfl_xor(mx, 16, 32));     // combine the two half-waves
        const float mnew  = fmaxf(mrun, mx);
        const float alpha = __builtin_amdgcn_exp2f(mrun - mnew);

        // exp2 + in-register C-layout -> A-layout conversion (coverage matches!)
        v16bf aP0, aP1;
        float rs = 0.f;
        #pragma unroll
        for (int v = 0; v < 8; ++v) {
            const float p0 = __builtin_amdgcn_exp2f(St0[v] - mnew);
            const float p1 = __builtin_amdgcn_exp2f(St1[v] - mnew);
            const float p2 = __builtin_amdgcn_exp2f(St2[v] - mnew);
            const float p3 = __builtin_amdgcn_exp2f(St3[v] - mnew);
            rs += (p0 + p1) + (p2 + p3);
            aP0[v] = (__bf16)p0;  aP0[v + 8] = (__bf16)p1;   // keys j0..j0+31
            aP1[v] = (__bf16)p2;  aP1[v + 8] = (__bf16)p3;   // keys j0+32..j0+63
        }
        rs += __shfl_xor(rs, 16, 32);
        lrun = lrun * alpha + rs;
        mrun = mnew;

        // rescale O: row r=v+8*hi needs alpha of query r (held in lane r)
        #pragma unroll
        for (int v = 0; v < 8; ++v) {
            const float ar = __shfl(alpha, v + rowln, 32);
            O0[v] *= ar;
            O1[v] *= ar;
        }
        O0 = wmma_bf16(aP0, bV00, O0);
        O0 = wmma_bf16(aP1, bV01, O0);
        O1 = wmma_bf16(aP0, bV10, O1);
        O1 = wmma_bf16(aP1, bV11, O1);
    }

    // Finalize: divide each query row by its sum, store Hid[b][head*32+d][token]
    const float linv = 1.0f / lrun;
    float* Hb = Hid + ((size_t)b * CCH + head * DH) * N_TOK;
    #pragma unroll
    for (int v = 0; v < 8; ++v) {
        const float lr = __shfl(linv, v + rowln, 32);
        const int tok  = qtile * 16 + v + 8 * hi;
        Hb[(size_t)(m)      * N_TOK + tok] = O0[v] * lr;
        Hb[(size_t)(m + 16) * N_TOK + tok] = O1[v] * lr;
    }
}

// ---------------------------------------------------------------------------
// Stage 3: output 1x1-conv projection + bias (f32)
// ---------------------------------------------------------------------------
__global__ void __launch_bounds__(256) out_proj_kernel(
    const float* __restrict__ Hid, const float* __restrict__ w,
    const float* __restrict__ bias, float* __restrict__ out)
{
    const int i = blockIdx.x * 256 + threadIdx.x;
    const int o = blockIdx.y;                       // 0..127
    const int b = blockIdx.z;
    const float* hb = Hid + (size_t)b * CCH * N_TOK + i;
    const float* wr = w + o * CCH;
    float acc = bias[o];
    #pragma unroll 8
    for (int c = 0; c < CCH; ++c)
        acc = fmaf(wr[c], hb[(size_t)c * N_TOK], acc);
    out[((size_t)b * CCH + o) * N_TOK + i] = acc;
}

// ---------------------------------------------------------------------------
extern "C" void kernel_launch(void* const* d_in, const int* in_sizes, int n_in,
                              void* d_out, int out_size, void* d_ws, size_t ws_size,
                              hipStream_t stream)
{
    const float* x     = (const float*)d_in[0];   // [4,128,64,64]
    const float* w_qkv = (const float*)d_in[1];   // [384,128]
    const float* w_out = (const float*)d_in[2];   // [128,128]
    const float* b_out = (const float*)d_in[3];   // [128]
    float* out = (float*)d_out;                   // [4,128,64,64] f32

    char* ws = (char*)d_ws;
    __bf16* Q   = (__bf16*)(ws);                       // 4 MB
    __bf16* K   = (__bf16*)(ws + ((size_t)4  << 20));  // 4 MB
    __bf16* Vt  = (__bf16*)(ws + ((size_t)8  << 20));  // 4 MB
    float*  Hid = (float*) (ws + ((size_t)12 << 20));  // 8 MB

    qkv_kernel<<<dim3(N_TOK / 256, 3 * CCH, NB), 256, 0, stream>>>(x, w_qkv, Q, K, Vt);
    flash_attn_kernel<<<dim3((NB * NH * (N_TOK / 16)) / 8), 256, 0, stream>>>(Q, K, Vt, Hid);
    out_proj_kernel<<<dim3(N_TOK / 256, CCH, NB), 256, 0, stream>>>(Hid, w_out, b_out, out);
}
